// CrossAttention_1_3384434230010
// MI455X (gfx1250) — matrix operand order repaired and checked by reading, not compiled, not run on hardware
//
#include <hip/hip_runtime.h>
#include <hip/hip_bf16.h>

#define HEADS    16
#define DIM_HEAD 64
#define BATCH    4
#define SEQ      2048
#define DMODEL   1024
#define INNER    1024
#define ATT_SCALE 0.125f   // 1/sqrt(64), folded into Q projection

typedef __attribute__((ext_vector_type(16))) __bf16 v16bf;
typedef __attribute__((ext_vector_type(8)))  __bf16 v8bf;
typedef __attribute__((ext_vector_type(8)))  float  v8f;

union BFrag { v16bf v; v8bf h[2]; };

__device__ inline v8f zero8() {
  v8f z;
#pragma unroll
  for (int i = 0; i < 8; ++i) z[i] = 0.0f;
  return z;
}

// CDNA5 async DMA: global -> LDS (16B), tracked by ASYNCcnt.
__device__ inline void async_load_b128(void* lds, const void* gptr) {
  unsigned la = (unsigned)(size_t)lds;                 // LDS offset (addr[31:0])
  unsigned long long ga = (unsigned long long)(size_t)gptr;
  asm volatile("global_load_async_to_lds_b128 %0, %1, off"
               :: "v"(la), "v"(ga) : "memory");
}
__device__ inline void wait_async0() {
  asm volatile("s_wait_asynccnt 0x0" ::: "memory");
}

// ---------------------------------------------------------------- converts
__global__ void cvt_f32_bf16(const float* __restrict__ src,
                             __bf16* __restrict__ dst, int n) {
  int i = blockIdx.x * blockDim.x + threadIdx.x;
  if (i < n) dst[i] = (__bf16)src[i];
}

// src is [K][N] row-major f32; dst is [N][K] bf16 (transposed)
__global__ void cvt_transpose(const float* __restrict__ src,
                              __bf16* __restrict__ dst, int K, int N) {
  int i = blockIdx.x * blockDim.x + threadIdx.x;
  if (i < K * N) {
    int k = i / N, n = i - k * N;
    dst[(size_t)n * K + k] = (__bf16)src[i];
  }
}

// ---------------------------------------------------------------- GEMM
// C[M,N] = A[M,K] (bf16, row-major) @ BT[N,K]^T (bf16, W^T layout)
// Block tile 128x128, BK=32, 8 waves (2x4), wave tile 64x32 (4x2 WMMA tiles).
// Double-buffered LDS fed by async global->LDS DMA.
// MODE 0: bf16 out (scaled)      -> Cb[row*N+col]
// MODE 1: bf16 out, V-transposed -> Cb[((b*H+h)*64+d)*SEQ + i]
// MODE 2: f32 out + bias         -> Cf[row*N+col]
template <int MODE>
__global__ void __launch_bounds__(256)
gemm_bf16(const __bf16* __restrict__ A, const __bf16* __restrict__ BT,
          __bf16* __restrict__ Cb, float* __restrict__ Cf,
          const float* __restrict__ bias, float outScale, int M, int N, int K) {
  __shared__ __bf16 sA[2][128][40];   // row stride 80B (16B multiple)
  __shared__ __bf16 sB[2][128][40];   // sB[.][n][k]

  const int tid    = threadIdx.x;
  const int lane   = tid & 31;
  const int w      = tid >> 5;
  const int wm     = (w >> 2) * 64;   // 0 / 64
  const int wn     = (w & 3) * 32;    // 0..96
  const int lanelo = lane & 15;
  const int hi     = lane >> 4;       // 0/1
  const int blockM = blockIdx.y * 128;
  const int blockN = blockIdx.x * 128;

  v8f acc[4][2];
#pragma unroll
  for (int i = 0; i < 4; ++i)
#pragma unroll
    for (int j = 0; j < 2; ++j) acc[i][j] = zero8();

  // async tile stage: 512 chunks of 8 halves per matrix, 2 chunks/thread each
  auto issue_tile = [&](int k0, int buf) {
#pragma unroll
    for (int j = 0; j < 2; ++j) {
      int ci  = tid + j * 256;
      int row = ci >> 2;
      int ko  = (ci & 3) * 8;
      async_load_b128(&sA[buf][row][ko],
                      A + (size_t)(blockM + row) * K + k0 + ko);
      async_load_b128(&sB[buf][row][ko],
                      BT + (size_t)(blockN + row) * K + k0 + ko);
    }
  };

  const int nk = K / 32;
  issue_tile(0, 0);

  for (int i = 0; i < nk; ++i) {
    const int buf = i & 1;
    wait_async0();          // own async writes for tile i complete
    __syncthreads();        // all waves' tile-i writes visible
    if (i + 1 < nk) issue_tile((i + 1) * 32, buf ^ 1);

    BFrag bf[2];
#pragma unroll
    for (int ns = 0; ns < 2; ++ns) {
      const __bf16* p = &sB[buf][wn + ns * 16 + lanelo][hi * 16];
      bf[ns].h[0] = *(const v8bf*)p;
      bf[ns].h[1] = *(const v8bf*)(p + 8);
    }
#pragma unroll
    for (int ms = 0; ms < 4; ++ms) {
      BFrag af;
      const __bf16* p = &sA[buf][wm + ms * 16 + lanelo][hi * 16];
      af.h[0] = *(const v8bf*)p;
      af.h[1] = *(const v8bf*)(p + 8);
#pragma unroll
      for (int ns = 0; ns < 2; ++ns) {
        acc[ms][ns] = __builtin_amdgcn_wmma_f32_16x16x32_bf16(
            false, af.v, false, bf[ns].v, (short)0, acc[ms][ns], false, false);
      }
    }
    __syncthreads();        // done reading buf before it is overwritten
  }

#pragma unroll
  for (int ms = 0; ms < 4; ++ms)
#pragma unroll
    for (int ns = 0; ns < 2; ++ns)
#pragma unroll
      for (int r = 0; r < 8; ++r) {
        int row = blockM + wm + ms * 16 + r + hi * 8;
        int col = blockN + wn + ns * 16 + lanelo;
        float v = acc[ms][ns][r];
        if (MODE == 2) {
          Cf[(size_t)row * N + col] = v + bias[col];
        } else if (MODE == 1) {
          // row = b*SEQ + i ; col = h*64 + d  ->  [(b*H+h)*64+d][i]
          int bb = row >> 11, ii = row & (SEQ - 1);
          int hh = col >> 6,  dd = col & 63;
          Cb[((size_t)((bb * HEADS + hh) * DIM_HEAD + dd) << 11) + ii] =
              (__bf16)v;
        } else {
          Cb[(size_t)row * N + col] = (__bf16)(v * outScale);
        }
      }
}

// ---------------------------------------------------------------- flash attention
// grid: (SEQ/128, BATCH*HEADS); 8 waves, each wave owns 16 query rows.
// Q pre-scaled by 1/sqrt(d). No online max (scores bounded << 88); row sums
// accumulated across all key blocks inside a WMMA C register via ones-matrix.
// K/V tiles double-buffered via async global->LDS DMA.
__global__ void __launch_bounds__(256)
flash_attn(const __bf16* __restrict__ Q, const __bf16* __restrict__ Kb,
           const __bf16* __restrict__ VT, __bf16* __restrict__ O) {
  __shared__ __bf16 sK[2][32][64];     // [key][d]  == BT layout for Q@K^T
  __shared__ __bf16 sVT[2][64][32];    // [d][key]  == BT layout for P@V
  __shared__ __bf16 sP[8][16][32];     // per-wave P tile

  const int tid    = threadIdx.x;
  const int lane   = tid & 31;
  const int w      = tid >> 5;
  const int lanelo = lane & 15;
  const int hi     = lane >> 4;
  const int b      = blockIdx.y / HEADS;
  const int h      = blockIdx.y % HEADS;
  const int bh     = blockIdx.y;
  const int i0     = blockIdx.x * 128 + w * 16;

  // resident Q fragments (16 rows x 64 d = two K-steps)
  BFrag qa[2];
  {
    const __bf16* qrow =
        Q + (size_t)(b * SEQ + i0 + lanelo) * INNER + h * DIM_HEAD;
#pragma unroll
    for (int t = 0; t < 2; ++t) {
      int d0 = t * 32 + hi * 16;
      qa[t].h[0] = *(const v8bf*)(qrow + d0);
      qa[t].h[1] = *(const v8bf*)(qrow + d0 + 8);
    }
  }

  // constant ones B-fragment (32x16 ones matrix) for row-sum WMMA
  BFrag ones;
#pragma unroll
  for (int e = 0; e < 16; ++e) ones.v[e] = (__bf16)1.0f;

  v8f o[4];
#pragma unroll
  for (int g = 0; g < 4; ++g) o[g] = zero8();
  v8f lacc = zero8();   // running row-sum of probabilities (WMMA-accumulated)

  // cooperative tile-load mapping
  const int jload  = tid >> 3;        // 0..31 (key row for K tile)
  const int dloadK = (tid & 7) * 8;   // 0..56
  const int dload  = tid >> 2;        // 0..63 (d row for VT tile)
  const int jloadV = (tid & 3) * 8;   // 0..24

  const __bf16* kbase  = Kb + (size_t)(b * SEQ) * INNER + h * DIM_HEAD;
  const __bf16* vtbase = VT + ((size_t)(bh * DIM_HEAD) << 11);  // [64][SEQ]

  auto issue_kv = [&](int kb, int buf) {
    async_load_b128(&sK[buf][jload][dloadK],
                    kbase + (size_t)(kb + jload) * INNER + dloadK);
    async_load_b128(&sVT[buf][dload][jloadV],
                    vtbase + ((size_t)dload << 11) + kb + jloadV);
  };

  const int nkb = SEQ / 32;
  issue_kv(0, 0);

  for (int i = 0; i < nkb; ++i) {
    const int buf = i & 1;
    wait_async0();
    __syncthreads();
    if (i + 1 < nkb) issue_kv((i + 1) * 32, buf ^ 1);

    // S = Q K^T (scale already folded into Q)
    v8f s[2];
#pragma unroll
    for (int g = 0; g < 2; ++g) {
      v8f a = zero8();
#pragma unroll
      for (int t = 0; t < 2; ++t) {
        const __bf16* p = &sK[buf][g * 16 + lanelo][t * 32 + hi * 16];
        BFrag bk;
        bk.h[0] = *(const v8bf*)p;
        bk.h[1] = *(const v8bf*)(p + 8);
        a = __builtin_amdgcn_wmma_f32_16x16x32_bf16(
            false, qa[t].v, false, bk.v, (short)0, a, false, false);
      }
      s[g] = a;
    }

    // P = exp(S)  (shift-free: |S| bounded far below exp overflow)
#pragma unroll
    for (int r = 0; r < 8; ++r) {
      sP[w][r + hi * 8][lanelo]      = (__bf16)__expf(s[0][r]);
      sP[w][r + hi * 8][16 + lanelo] = (__bf16)__expf(s[1][r]);
    }

    // reload P as A-fragment (per-wave LDS region; wave-order DS keeps RAW)
    BFrag pa;
    {
      const __bf16* p = &sP[w][lanelo][hi * 16];
      pa.h[0] = *(const v8bf*)p;
      pa.h[1] = *(const v8bf*)(p + 8);
    }

    // running row-sum: lacc += P @ ones
    lacc = __builtin_amdgcn_wmma_f32_16x16x32_bf16(
        false, pa.v, false, ones.v, (short)0, lacc, false, false);

    // O += P @ V
#pragma unroll
    for (int g = 0; g < 4; ++g) {
      const __bf16* p = &sVT[buf][g * 16 + lanelo][hi * 16];
      BFrag bv;
      bv.h[0] = *(const v8bf*)p;
      bv.h[1] = *(const v8bf*)(p + 8);
      o[g] = __builtin_amdgcn_wmma_f32_16x16x32_bf16(
          false, pa.v, false, bv.v, (short)0, o[g], false, false);
    }
    __syncthreads();   // done reading buf before DMA overwrites it
  }

  // normalize and store merged-head attention output (bf16)
#pragma unroll
  for (int r = 0; r < 8; ++r) {
    float inv = 1.0f / lacc[r];
    int row = i0 + r + hi * 8;
    __bf16* orow = O + (size_t)(b * SEQ + row) * INNER + h * DIM_HEAD;
#pragma unroll
    for (int g = 0; g < 4; ++g)
      orow[g * 16 + lanelo] = (__bf16)(o[g][r] * inv);
  }
}

// ---------------------------------------------------------------- launch
extern "C" void kernel_launch(void* const* d_in, const int* in_sizes, int n_in,
                              void* d_out, int out_size, void* d_ws,
                              size_t ws_size, hipStream_t stream) {
  const float* x1  = (const float*)d_in[0];
  const float* x2  = (const float*)d_in[1];
  const float* Wq  = (const float*)d_in[2];
  const float* Wkv = (const float*)d_in[3];
  const float* Wo  = (const float*)d_in[4];
  const float* bo  = (const float*)d_in[5];
  float* out = (float*)d_out;

  char* ws = (char*)d_ws;
  size_t off = 0;
  auto alloc = [&](size_t bytes) -> void* {
    void* p = ws + off;
    off += (bytes + 255) & ~(size_t)255;
    return p;
  };
  const size_t M = (size_t)BATCH * SEQ;                 // 8192
  __bf16* x1b  = (__bf16*)alloc(M * DMODEL * 2);
  __bf16* x2b  = (__bf16*)alloc(M * DMODEL * 2);
  __bf16* wqT  = (__bf16*)alloc((size_t)DMODEL * INNER * 2);
  __bf16* wkvT = (__bf16*)alloc((size_t)DMODEL * 2 * INNER * 2);
  __bf16* woT  = (__bf16*)alloc((size_t)INNER * DMODEL * 2);
  __bf16* qb   = (__bf16*)alloc(M * INNER * 2);
  __bf16* kb   = (__bf16*)alloc(M * INNER * 2);
  __bf16* vtb  = (__bf16*)alloc(M * INNER * 2);   // [b,h,d,seq]
  __bf16* aob  = (__bf16*)alloc(M * INNER * 2);

  int nx = (int)(M * DMODEL);
  cvt_f32_bf16<<<(nx + 255) / 256, 256, 0, stream>>>(x1, x1b, nx);
  cvt_f32_bf16<<<(nx + 255) / 256, 256, 0, stream>>>(x2, x2b, nx);
  int nw = DMODEL * INNER;
  cvt_transpose<<<(nw + 255) / 256, 256, 0, stream>>>(Wq, wqT, DMODEL, INNER);
  cvt_transpose<<<(2 * nw + 255) / 256, 256, 0, stream>>>(Wkv, wkvT, DMODEL,
                                                          2 * INNER);
  cvt_transpose<<<(nw + 255) / 256, 256, 0, stream>>>(Wo, woT, INNER, DMODEL);

  dim3 g1(INNER / 128, (int)(M / 128));
  // q = (x1 @ Wq) * 1/sqrt(d)
  gemm_bf16<0><<<g1, 256, 0, stream>>>(x1b, wqT, qb, nullptr, nullptr,
                                       ATT_SCALE, (int)M, INNER, DMODEL);
  // k = x2 @ Wkv[:, :1024]
  gemm_bf16<0><<<g1, 256, 0, stream>>>(x2b, wkvT, kb, nullptr, nullptr, 1.0f,
                                       (int)M, INNER, DMODEL);
  // vT = (x2 @ Wkv[:, 1024:]) stored [b,h,d,seq]
  gemm_bf16<1><<<g1, 256, 0, stream>>>(x2b, wkvT + (size_t)INNER * DMODEL, vtb,
                                       nullptr, nullptr, 1.0f, (int)M, INNER,
                                       DMODEL);
  // attention
  flash_attn<<<dim3(SEQ / 128, BATCH * HEADS), 256, 0, stream>>>(qb, kb, vtb,
                                                                 aob);
  // out = ao @ Wo + bo  (f32 output)
  gemm_bf16<2><<<dim3(DMODEL / 128, (int)(M / 128)), 256, 0, stream>>>(
      aob, woT, nullptr, out, bo, 1.0f, (int)M, DMODEL, INNER);
}
